// RestoreNet_90228672954714
// MI455X (gfx1250) — matrix operand order repaired and checked by reading, not compiled, not run on hardware
//
#include <hip/hip_runtime.h>
#include <hip/hip_bf16.h>

typedef __bf16 bf16_t;
typedef __attribute__((ext_vector_type(16))) __bf16 v16bf;
typedef __attribute__((ext_vector_type(8)))  __bf16 v8bf;
typedef __attribute__((ext_vector_type(8)))  float  v8f;

union V8F   { v8f v; float f[8]; };
union V16BF { v16bf v; v8bf h[2]; };

// ---- WMMA fragment loaders (16-bit operands; contiguous k order, same for A and B) ----
// A 16x32 bf16: lanes 0-15 hold row M=lane, K=0..15; lanes 16-31 same row, K=16..31
__device__ __forceinline__ v16bf load_a_frag(const bf16_t* rowk, int hi) {
  const v8bf* p = (const v8bf*)rowk;     // rowk = row base + kstep*32 (16B aligned)
  V16BF r; r.h[0] = p[hi ? 2 : 0]; r.h[1] = p[hi ? 3 : 1];
  return r.v;
}
// B 32x16 bf16: lane holds column N=lane&15, K chunk = (lane>=16 ? 16 : 0) + 0..15 contiguous
__device__ __forceinline__ v16bf load_b_frag(const bf16_t* colk) {
  const v8bf* p = (const v8bf*)colk;     // colk = Bt[col]*KP + kstep*32 + (hi?16:0)
  V16BF r; r.h[0] = p[0]; r.h[1] = p[1];
  return r.v;
}

// =====================================================================
// Fused pairwise-score GEMM + per-query top-k (KNN / cosine aggregation)
// A = candidate features (M), B = query features (N): each lane owns one
// query column; per-lane sorted top-k merged across the two lane halves.
// Candidate tiles are streamed once; prefetch the next tile so L2->WGP
// latency overlaps the WMMAs (global_prefetch_b8).
// =====================================================================
template<int CSTRIDE, int KSEL, bool LARGEST, bool USE_SQ>
__global__ void knn_topk_kernel(const bf16_t* __restrict__ xc, const bf16_t* __restrict__ xq,
                                const float* __restrict__ sqc, const float* __restrict__ sqq,
                                int* __restrict__ idx_out, int npts)
{
  constexpr int NK = CSTRIDE / 32;
  const int b = blockIdx.y, qt = blockIdx.x, lane = threadIdx.x;
  const int n = lane & 15, hi = lane >> 4, moff = hi * 8;

  const size_t cbase = (size_t)b * npts * CSTRIDE;
  const bf16_t* xqB = xq + cbase + (size_t)(qt * 16 + n) * CSTRIDE + hi * 16;
  v16bf bq[NK];
  for (int ks = 0; ks < NK; ks++) bq[ks] = load_b_frag(xqB + ks * 32);
  const float sq_q = USE_SQ ? sqq[(size_t)b * npts + qt * 16 + n] : 0.f;

  float bd[KSEL]; int bix[KSEL];
  for (int i = 0; i < KSEL; i++) { bd[i] = LARGEST ? -1e30f : 1e30f; bix[i] = 0; }

  const int ntiles = npts / 16;
  for (int ct = 0; ct < ntiles; ct++) {
    const bf16_t* xcT = xc + cbase + (size_t)(ct * 16 + n) * CSTRIDE;
    if (ct + 1 < ntiles) {
      // prefetch next candidate tile row for this lane (one cacheline probe)
      __builtin_prefetch(xcT + (size_t)16 * CSTRIDE, 0, 3);
    }
    V8F acc; for (int v = 0; v < 8; v++) acc.f[v] = 0.f;
    for (int ks = 0; ks < NK; ks++) {
      v16bf a = load_a_frag(xcT + ks * 32, hi);
      acc.v = __builtin_amdgcn_wmma_f32_16x16x32_bf16(false, a, false, bq[ks],
                                                      (short)0, acc.v, false, false);
    }
    const float* sp = USE_SQ ? (sqc + (size_t)b * npts + ct * 16 + moff) : nullptr;
    for (int v = 0; v < 8; v++) {
      int cand = ct * 16 + moff + v;
      float s = USE_SQ ? (sq_q + sp[v] - 2.f * acc.f[v]) : acc.f[v];
      if (LARGEST) {
        if (s > bd[KSEL - 1]) {
          int j = KSEL - 1;
          while (j > 0 && bd[j - 1] < s) { bd[j] = bd[j - 1]; bix[j] = bix[j - 1]; j--; }
          bd[j] = s; bix[j] = cand;
        }
      } else {
        if (s < bd[KSEL - 1]) {
          int j = KSEL - 1;
          while (j > 0 && bd[j - 1] > s) { bd[j] = bd[j - 1]; bix[j] = bix[j - 1]; j--; }
          bd[j] = s; bix[j] = cand;
        }
      }
    }
  }

  __shared__ float sd[16][2][KSEL];
  __shared__ int   si[16][2][KSEL];
  for (int i = 0; i < KSEL; i++) { sd[n][hi][i] = bd[i]; si[n][hi][i] = bix[i]; }
  __syncthreads();
  if (lane < 16) {
    int a = 0, c = 0;
    int* op = idx_out + ((size_t)b * npts + qt * 16 + lane) * KSEL;
    for (int j = 0; j < KSEL; j++) {
      bool useA;
      if (a >= KSEL) useA = false;
      else if (c >= KSEL) useA = true;
      else useA = LARGEST ? (sd[lane][0][a] >= sd[lane][1][c])
                          : (sd[lane][0][a] <= sd[lane][1][c]);
      if (useA) { op[j] = si[lane][0][a]; a++; }
      else      { op[j] = si[lane][1][c]; c++; }
    }
  }
}

// =====================================================================
// EdgeConv: h = max_k leakyrelu(cat(x_j - x_i, x_i) @ W + b)
// 4 waves x 16 output channels; A tile (16 pts x KPAD) staged in LDS,
// weight B-fragments held in registers across all 20 neighbor slots.
// =====================================================================
template<int CIN, int KPAD>
__global__ void edge_conv_kernel(const float* __restrict__ x, const int* __restrict__ idx,
                                 const bf16_t* __restrict__ Wt /*[64][KPAD]*/,
                                 const float* __restrict__ bias,
                                 float* __restrict__ out /*[B,N,64]*/, int npts)
{
  constexpr int NK = KPAD / 32;
  const int b = blockIdx.y, nt = blockIdx.x;
  const int tid = threadIdx.x, wave = tid >> 5, lane = tid & 31;
  const int n = lane & 15, hi = lane >> 4, moff = hi * 8;
  const int col = wave * 16 + n;

  __shared__ __align__(16) bf16_t As[16 * KPAD];

  v16bf breg[NK];
  for (int ks = 0; ks < NK; ks++)
    breg[ks] = load_b_frag(Wt + (size_t)col * KPAD + ks * 32 + hi * 16);
  const float bb = bias[col];

  float mx[8];
  for (int v = 0; v < 8; v++) mx[v] = -1e30f;

  const float* xB = x + (size_t)b * npts * CIN;
  for (int k = 0; k < 20; k++) {
    for (int e = tid; e < 16 * KPAD; e += blockDim.x) {
      int r = e / KPAD, c = e % KPAD;
      int pi = nt * 16 + r;
      int pj = idx[((size_t)b * npts + pi) * 20 + k];
      float val = 0.f;
      if (c < CIN)          val = xB[(size_t)pj * CIN + c] - xB[(size_t)pi * CIN + c];
      else if (c < 2 * CIN) val = xB[(size_t)pi * CIN + (c - CIN)];
      As[e] = (bf16_t)val;
    }
    __syncthreads();
    V8F acc; for (int v = 0; v < 8; v++) acc.f[v] = 0.f;
    for (int ks = 0; ks < NK; ks++) {
      v16bf a = load_a_frag(As + n * KPAD + ks * 32, hi);
      acc.v = __builtin_amdgcn_wmma_f32_16x16x32_bf16(false, a, false, breg[ks],
                                                      (short)0, acc.v, false, false);
    }
    for (int v = 0; v < 8; v++) {
      float h = acc.f[v] + bb;
      h = h > 0.f ? h : 0.2f * h;
      mx[v] = fmaxf(mx[v], h);
    }
    __syncthreads();
  }
  for (int v = 0; v < 8; v++)
    out[((size_t)b * npts + nt * 16 + v + moff) * 64 + col] = mx[v];
}

// =====================================================================
// Generic GEMM: C[M,N] = act(A[M,KP](bf16) @ Bt[N,KP]^T (bf16) + bias)
// =====================================================================
__global__ void gemm_bt_kernel(const bf16_t* __restrict__ A, const bf16_t* __restrict__ Bt,
                               const float* __restrict__ bias, float* __restrict__ C,
                               int M, int N, int KP, int lrelu)
{
  const int mt = blockIdx.x;
  const int tid = threadIdx.x, wave = tid >> 5, lane = tid & 31;
  const int n = lane & 15, hi = lane >> 4, moff = hi * 8;
  const int col = blockIdx.y * 64 + wave * 16 + n;
  const int NK = KP / 32;

  V8F acc; for (int v = 0; v < 8; v++) acc.f[v] = 0.f;
  const bf16_t* Arow = A + (size_t)(mt * 16 + n) * KP;
  const bf16_t* Bcol = Bt + (size_t)col * KP + hi * 16;
  __builtin_prefetch(Arow, 0, 3);   // warm A row (weights stay L2-hot)
  for (int ks = 0; ks < NK; ks++) {
    v16bf a = load_a_frag(Arow + ks * 32, hi);
    v16bf bb = load_b_frag(Bcol + ks * 32);
    acc.v = __builtin_amdgcn_wmma_f32_16x16x32_bf16(false, a, false, bb,
                                                    (short)0, acc.v, false, false);
  }
  float bv = bias ? bias[col] : 0.f;
  for (int v = 0; v < 8; v++) {
    float h = acc.f[v] + bv;
    if (lrelu) h = h > 0.f ? h : 0.2f * h;
    C[((size_t)mt * 16 + v + moff) * N + col] = h;
  }
}

// =====================================================================
// Farthest point sampling: one workgroup per batch, dist array in LDS.
// =====================================================================
__global__ void fps_kernel(const float* __restrict__ s1, int n1,
                           const float* __restrict__ s2, int n2,
                           float* __restrict__ out, int npoint)
{
  const int b = blockIdx.x, tid = threadIdx.x;
  const int ntot = n1 + n2;
  __shared__ float dist[4096];
  __shared__ float rv[256];
  __shared__ int   ri[256];
  __shared__ float cen[3];
  __shared__ int   farS;

  for (int i = tid; i < ntot; i += 256) dist[i] = 1e10f;
  if (tid == 0) farS = 0;
  __syncthreads();

  for (int t = 0; t < npoint; t++) {
    int far = farS;
    if (tid == 0) {
      const float* pp = (far < n1) ? (s1 + ((size_t)b * n1 + far) * 3)
                                   : (s2 + ((size_t)b * n2 + (far - n1)) * 3);
      for (int c = 0; c < 3; c++) {
        float v = pp[c]; cen[c] = v;
        out[((size_t)b * npoint + t) * 3 + c] = v;
      }
    }
    __syncthreads();
    float cx = cen[0], cy = cen[1], cz = cen[2];
    float bv = -1.f; int bi = 0;
    for (int i = tid; i < ntot; i += 256) {
      const float* pp = (i < n1) ? (s1 + ((size_t)b * n1 + i) * 3)
                                 : (s2 + ((size_t)b * n2 + (i - n1)) * 3);
      float dx = pp[0] - cx, dy = pp[1] - cy, dz = pp[2] - cz;
      float d = dx * dx + dy * dy + dz * dz;
      float nd = fminf(dist[i], d);
      dist[i] = nd;
      if (nd > bv) { bv = nd; bi = i; }
    }
    rv[tid] = bv; ri[tid] = bi;
    __syncthreads();
    for (int s = 128; s > 0; s >>= 1) {
      if (tid < s && rv[tid + s] > rv[tid]) { rv[tid] = rv[tid + s]; ri[tid] = ri[tid + s]; }
      __syncthreads();
    }
    if (tid == 0) farS = ri[0];
    __syncthreads();
  }
}

// ---------------- elementwise / prep kernels ----------------
__global__ void pad_rows_bf16_kernel(const float* __restrict__ src, bf16_t* __restrict__ dst,
                                     long R, int Cs, int Cd) {
  long e = (long)blockIdx.x * blockDim.x + threadIdx.x;
  if (e >= R * Cd) return;
  long r = e / Cd; int c = (int)(e - r * Cd);
  dst[e] = (bf16_t)(c < Cs ? src[r * Cs + c] : 0.f);
}
__global__ void transpose_pad_bf16_kernel(const float* __restrict__ W /*[K,N]*/,
                                          bf16_t* __restrict__ Wt /*[N,KP]*/,
                                          int K, int N, int KP) {
  int e = blockIdx.x * blockDim.x + threadIdx.x;
  if (e >= N * KP) return;
  int nn = e / KP, kp = e - nn * KP;
  Wt[e] = (bf16_t)(kp < K ? W[(size_t)kp * N + nn] : 0.f);
}
__global__ void sqnorm_kernel(const float* __restrict__ x, float* __restrict__ sq, long R, int C) {
  long r = (long)blockIdx.x * blockDim.x + threadIdx.x;
  if (r >= R) return;
  float s = 0.f;
  for (int c = 0; c < C; c++) { float v = x[r * C + c]; s += v * v; }
  sq[r] = s;
}
__global__ void normalize_bf16_kernel(const float* __restrict__ x, bf16_t* __restrict__ dst,
                                      long R, int C) {
  long r = (long)blockIdx.x * blockDim.x + threadIdx.x;
  if (r >= R) return;
  float s = 0.f;
  for (int c = 0; c < C; c++) { float v = x[r * C + c]; s += v * v; }
  float rs = rsqrtf(s + 1e-8f);
  for (int c = 0; c < C; c++) dst[r * C + c] = (bf16_t)(x[r * C + c] * rs);
}
__global__ void agg_pool_kernel(const float* __restrict__ ffp, const int* __restrict__ aidx,
                                float* __restrict__ fagg, int npts) {
  int b = blockIdx.y, nq = blockIdx.x, ch = threadIdx.x;  // blockDim = 64
  const int* ip = aidx + ((size_t)b * npts + nq) * 8;
  float mean = 0.f, mx = -1e30f;
  for (int j = 0; j < 8; j++) {
    float v = ffp[((size_t)b * npts + ip[j]) * 64 + ch];
    mean += v; mx = fmaxf(mx, v);
  }
  float* o = fagg + ((size_t)b * npts + nq) * 128;
  o[ch] = mean * 0.125f; o[64 + ch] = mx;
}
__global__ void build_f_kernel(const float* __restrict__ ffq, const float* __restrict__ fagg,
                               const float* __restrict__ qs, bf16_t* __restrict__ fpad) {
  long r = blockIdx.x; int c = threadIdx.x;  // blockDim = 224
  float v = 0.f;
  if (c < 64)       v = ffq[r * 64 + c];
  else if (c < 192) v = fagg[r * 128 + (c - 64)];
  else if (c < 195) v = qs[r * 3 + (c - 192)];
  fpad[r * 224 + c] = (bf16_t)v;
}
__global__ void cvt_bf16_kernel(const float* __restrict__ s, bf16_t* __restrict__ d, long n) {
  long e = (long)blockIdx.x * blockDim.x + threadIdx.x;
  if (e < n) d[e] = (bf16_t)s[e];
}
__global__ void last_layer_kernel(const float* __restrict__ h2, const float* __restrict__ Wr3,
                                  const float* __restrict__ br3, const float* __restrict__ qs,
                                  float* __restrict__ qref, long total) {
  long r = (long)blockIdx.x * blockDim.x + threadIdx.x;
  if (r >= total) return;
  const float* h = h2 + r * 64;
  for (int c = 0; c < 3; c++) {
    float s = br3[c];
    for (int k = 0; k < 64; k++) s += h[k] * Wr3[k * 3 + c];
    qref[r * 3 + c] = qs[r * 3 + c] + s;
  }
}

// =====================================================================
extern "C" void kernel_launch(void* const* d_in, const int* in_sizes, int n_in,
                              void* d_out, int out_size, void* d_ws, size_t ws_size,
                              hipStream_t stream) {
  (void)in_sizes; (void)n_in; (void)out_size; (void)ws_size;
  constexpr int B = 8, NPT = 2048;
  constexpr long TOT = (long)B * NPT;  // 16384

  const float* p   = (const float*)d_in[0];
  const float* q   = (const float*)d_in[1];
  const float* W1  = (const float*)d_in[2];
  const float* b1  = (const float*)d_in[3];
  const float* W2  = (const float*)d_in[4];
  const float* b2  = (const float*)d_in[5];
  const float* Wr1 = (const float*)d_in[6];
  const float* br1 = (const float*)d_in[7];
  const float* Wr2 = (const float*)d_in[8];
  const float* br2 = (const float*)d_in[9];
  const float* Wr3 = (const float*)d_in[10];
  const float* br3 = (const float*)d_in[11];

  float* q_refine = (float*)d_out;               // [B,N,3]
  float* restored = (float*)d_out + TOT * 3;     // [B,N,3]

  // --- workspace bump allocator (256B aligned) ---
  char* wsp = (char*)d_ws; size_t off = 0;
  auto alloc = [&](size_t bytes) -> void* {
    off = (off + 255) & ~(size_t)255;
    void* ptr = wsp + off; off += bytes; return ptr;
  };
  float*  qs    = (float*) alloc(TOT * 3 * 4);
  bf16_t* W1t   = (bf16_t*)alloc(64 * 32 * 2);
  bf16_t* W2t   = (bf16_t*)alloc(64 * 128 * 2);
  bf16_t* Wr1t  = (bf16_t*)alloc(128 * 224 * 2);
  bf16_t* Wr2t  = (bf16_t*)alloc(64 * 128 * 2);
  bf16_t* xb_in = (bf16_t*)alloc(TOT * 32 * 2);
  float*  sq_in = (float*) alloc(TOT * 4);
  int*    idx   = (int*)   alloc(TOT * 20 * 4);
  float*  x1    = (float*) alloc(TOT * 64 * 4);
  bf16_t* xb1   = (bf16_t*)alloc(TOT * 64 * 2);
  float*  sq1   = (float*) alloc(TOT * 4);
  float*  ffp   = (float*) alloc(TOT * 64 * 4);
  float*  ffq   = (float*) alloc(TOT * 64 * 4);
  bf16_t* ffpn  = (bf16_t*)alloc(TOT * 64 * 2);
  bf16_t* ffqn  = (bf16_t*)alloc(TOT * 64 * 2);
  int*    aidx  = (int*)   alloc(TOT * 8 * 4);
  float*  fagg  = (float*) alloc(TOT * 128 * 4);
  bf16_t* fpad  = (bf16_t*)alloc(TOT * 224 * 2);
  float*  h1    = (float*) alloc(TOT * 128 * 4);
  bf16_t* h1b   = (bf16_t*)alloc(TOT * 128 * 2);
  float*  h2    = (float*) alloc(TOT * 64 * 4);

  auto cdiv = [](long a, long b) { return (unsigned)((a + b - 1) / b); };

  // qs = fps(concat(p, q))
  fps_kernel<<<B, 256, 0, stream>>>(p, NPT, q, NPT, qs, NPT);

  // weight prep (transpose + K-pad to bf16)
  transpose_pad_bf16_kernel<<<cdiv(64 * 32, 256), 256, 0, stream>>>(W1, W1t, 6, 64, 32);
  transpose_pad_bf16_kernel<<<cdiv(64 * 128, 256), 256, 0, stream>>>(W2, W2t, 128, 64, 128);
  transpose_pad_bf16_kernel<<<cdiv(128 * 224, 256), 256, 0, stream>>>(Wr1, Wr1t, 195, 128, 224);
  transpose_pad_bf16_kernel<<<cdiv(64 * 128, 256), 256, 0, stream>>>(Wr2, Wr2t, 128, 64, 128);

  // DGCNN encoder (shared scratch; run for p then qs)
  auto encode = [&](const float* pts, float* ffout) {
    pad_rows_bf16_kernel<<<cdiv(TOT * 32, 256), 256, 0, stream>>>(pts, xb_in, TOT, 3, 32);
    sqnorm_kernel<<<cdiv(TOT, 256), 256, 0, stream>>>(pts, sq_in, TOT, 3);
    knn_topk_kernel<32, 20, false, true><<<dim3(NPT / 16, B), 32, 0, stream>>>(
        xb_in, xb_in, sq_in, sq_in, idx, NPT);
    edge_conv_kernel<3, 32><<<dim3(NPT / 16, B), 128, 0, stream>>>(
        pts, idx, W1t, b1, x1, NPT);
    pad_rows_bf16_kernel<<<cdiv(TOT * 64, 256), 256, 0, stream>>>(x1, xb1, TOT, 64, 64);
    sqnorm_kernel<<<cdiv(TOT, 256), 256, 0, stream>>>(x1, sq1, TOT, 64);
    knn_topk_kernel<64, 20, false, true><<<dim3(NPT / 16, B), 32, 0, stream>>>(
        xb1, xb1, sq1, sq1, idx, NPT);
    edge_conv_kernel<64, 128><<<dim3(NPT / 16, B), 128, 0, stream>>>(
        x1, idx, W2t, b2, ffout, NPT);
  };
  encode(p, ffp);
  encode(qs, ffq);

  // aggregation: cosine top-8 over normalized features, then mean|max pool
  normalize_bf16_kernel<<<cdiv(TOT, 256), 256, 0, stream>>>(ffp, ffpn, TOT, 64);
  normalize_bf16_kernel<<<cdiv(TOT, 256), 256, 0, stream>>>(ffq, ffqn, TOT, 64);
  knn_topk_kernel<64, 8, true, false><<<dim3(NPT / 16, B), 32, 0, stream>>>(
      ffpn, ffqn, nullptr, nullptr, aidx, NPT);
  agg_pool_kernel<<<dim3(NPT, B), 64, 0, stream>>>(ffp, aidx, fagg, NPT);

  // refine MLP: 195(->224 pad) -> 128 -> 64 -> 3, residual onto qs
  build_f_kernel<<<(unsigned)TOT, 224, 0, stream>>>(ffq, fagg, qs, fpad);
  gemm_bt_kernel<<<dim3((unsigned)(TOT / 16), 2), 128, 0, stream>>>(
      fpad, Wr1t, br1, h1, (int)TOT, 128, 224, 1);
  cvt_bf16_kernel<<<cdiv(TOT * 128, 256), 256, 0, stream>>>(h1, h1b, TOT * 128);
  gemm_bt_kernel<<<dim3((unsigned)(TOT / 16), 1), 128, 0, stream>>>(
      h1b, Wr2t, br2, h2, (int)TOT, 64, 128, 1);
  last_layer_kernel<<<cdiv(TOT, 256), 256, 0, stream>>>(h2, Wr3, br3, qs, q_refine, TOT);

  // restored = fps(concat(q_refine, p))
  fps_kernel<<<B, 256, 0, stream>>>(q_refine, NPT, p, NPT, restored, NPT);
}
